// Merge_Attention_81939386073219
// MI455X (gfx1250) — matrix operand order repaired and checked by reading, not compiled, not run on hardware
//
#include <hip/hip_runtime.h>

#define HWSZ 9216
#define CDIM 256
#define NT   64
#define PAD  16
#define NTILES_PER_IMG (HWSZ / NT)   /* 144 */
#define BATCH 8
#define HEADS 8

typedef __bf16 v16bf __attribute__((ext_vector_type(16)));
typedef __bf16 v8bf  __attribute__((ext_vector_type(8)));
typedef float  v8f   __attribute__((ext_vector_type(8)));

union FragAB { v16bf v; v8bf h[2]; };

__device__ __forceinline__ v8f zero8() {
  v8f z = {0.f, 0.f, 0.f, 0.f, 0.f, 0.f, 0.f, 0.f};
  return z;
}

__device__ __forceinline__ v8f wmma_bf16(v16bf a, v16bf b, v8f c) {
  return __builtin_amdgcn_wmma_f32_16x16x32_bf16(
      /*neg_a=*/false, a, /*neg_b=*/false, b,
      /*c_mod=*/(short)0, c, /*reuse_a=*/false, /*reuse_b=*/false);
}

// ---------------------------------------------------------------------------
// Weight prep: Wqx = W_cat[:, :128] @ W_qr ; Wqy = W_cat[:,128:] @ W_qd (bf16)
//              Wkb/Wvb = bf16(W_kf/W_vf) ; bq = W_cat @ [b_qr;b_qd] + b_cat
// ---------------------------------------------------------------------------
__global__ __launch_bounds__(256)
void wprep_kernel(const float* __restrict__ Wqr, const float* __restrict__ bqr,
                  const float* __restrict__ Wqd, const float* __restrict__ bqd,
                  const float* __restrict__ Wkf, const float* __restrict__ Wvf,
                  const float* __restrict__ Wcat, const float* __restrict__ bcat,
                  __bf16* __restrict__ Wqx, __bf16* __restrict__ Wqy,
                  __bf16* __restrict__ Wkb, __bf16* __restrict__ Wvb,
                  float* __restrict__ bq)
{
  const int c = blockIdx.x;    // 0..255  (output row)
  const int k = threadIdx.x;   // 0..255  (inner dim)
  const float* wc = Wcat + (size_t)c * CDIM;
  float s1 = 0.f, s2 = 0.f;
  #pragma unroll 8
  for (int j = 0; j < 128; ++j) {
    s1 += wc[j]       * Wqr[(size_t)j * CDIM + k];
    s2 += wc[128 + j] * Wqd[(size_t)j * CDIM + k];
  }
  const size_t idx = (size_t)c * CDIM + k;
  Wqx[idx] = (__bf16)s1;
  Wqy[idx] = (__bf16)s2;
  Wkb[idx] = (__bf16)Wkf[idx];
  Wvb[idx] = (__bf16)Wvf[idx];

  __shared__ float red[256];
  const float bb = (k < 128) ? bqr[k] : bqd[k - 128];
  red[k] = wc[k] * bb;
  __syncthreads();
  for (int s = 128; s > 0; s >>= 1) {
    if (k < s) red[k] += red[k + s];
    __syncthreads();
  }
  if (k == 0) bq[c] = red[0] + bcat[c];
}

// ---------------------------------------------------------------------------
// Generic 256xN GEMM pass:  out[b,m,n] = Wa@inA (+ Wb@inB) + bias (+ residual)
// 8 waves/block, each wave owns a 32(M)x64(N) slab of the 256x64 output tile.
// Optionally emits per-tile sum-of-squares partials (deterministic).
// ---------------------------------------------------------------------------
__device__ __forceinline__ void mma_pass(const __bf16* __restrict__ W,
                                         const __bf16 (*lds)[CDIM + PAD],
                                         int wave, int lane, v8f (*acc)[4])
{
  const int mrow  = lane & 15;
  const int khalf = (lane >> 4) << 4;      // A/B-frag K half: contiguous k = 16h + s
  const __bf16* w0 = W + (size_t)(wave * 32 + mrow) * CDIM + khalf;
  const __bf16* w1 = w0 + (size_t)16 * CDIM;
  #pragma unroll
  for (int kb = 0; kb < CDIM; kb += 32) {
    FragAB a0, a1, bf[4];
    a0.v = *(const v16bf*)(w0 + kb);
    a1.v = *(const v16bf*)(w1 + kb);
    #pragma unroll
    for (int nt = 0; nt < 4; ++nt)
      bf[nt].v = *(const v16bf*)(&lds[nt * 16 + mrow][kb + khalf]);
    #pragma unroll
    for (int nt = 0; nt < 4; ++nt) {
      acc[0][nt] = wmma_bf16(a0.v, bf[nt].v, acc[0][nt]);
      acc[1][nt] = wmma_bf16(a1.v, bf[nt].v, acc[1][nt]);
    }
  }
}

__global__ __launch_bounds__(256)
void gemm256_kernel(const __bf16* __restrict__ Wa,   // [256,256] bf16 (or [B,256,256] if perBatchW)
                    const __bf16* __restrict__ Wb,   // optional 2nd weight
                    const float*  __restrict__ inAf, // fp32 input A, or
                    const __bf16* __restrict__ inAb, // bf16 input A
                    const float*  __restrict__ inBf, // fp32 input B (with Wb)
                    const float*  __restrict__ bias, // [256] or null
                    __bf16* __restrict__ outB,       // bf16 out, or
                    float*  __restrict__ outF,       // fp32 out
                    float*  __restrict__ sspart,     // [B*256,144] sumsq partials or null
                    const float* __restrict__ residual, // fp32 add (with outF)
                    int perBatchW)
{
  __shared__ __align__(32) __bf16 lds[NT][CDIM + PAD];

  const int blk  = blockIdx.x;
  const int b    = blk / NTILES_PER_IMG;
  const int tile = blk % NTILES_PER_IMG;
  const int n0   = tile * NT;
  const int tid  = threadIdx.x;
  const int wave = tid >> 5;
  const int lane = tid & 31;
  const size_t ioBase = (size_t)b * CDIM * HWSZ;

  // stage input-A tile into LDS as bf16, layout [n][k]
  {
    const int nl = tid & (NT - 1);
    for (int k = tid / NT; k < CDIM; k += 256 / NT) {
      const size_t gi = ioBase + (size_t)k * HWSZ + n0 + nl;
      lds[nl][k] = inAb ? inAb[gi] : (__bf16)inAf[gi];
    }
  }
  __syncthreads();

  v8f acc[2][4];
  #pragma unroll
  for (int mt = 0; mt < 2; ++mt)
    #pragma unroll
    for (int nt = 0; nt < 4; ++nt) acc[mt][nt] = zero8();

  const __bf16* WaP = Wa + (perBatchW ? (size_t)b * CDIM * CDIM : 0);
  mma_pass(WaP, lds, wave, lane, acc);

  if (Wb) {  // second accumulation pass (q = Wqx@x + Wqy@y)
    __syncthreads();
    const int nl = tid & (NT - 1);
    for (int k = tid / NT; k < CDIM; k += 256 / NT)
      lds[nl][k] = (__bf16)inBf[ioBase + (size_t)k * HWSZ + n0 + nl];
    __syncthreads();
    mma_pass(Wb, lds, wave, lane, acc);
  }

  // epilogue: bias, store, sum-of-squares partials
  const int ncol  = lane & 15;
  const int half8 = (lane >> 4) << 3;
  float ssl[2][8];
  #pragma unroll
  for (int mt = 0; mt < 2; ++mt)
    #pragma unroll
    for (int j = 0; j < 8; ++j) ssl[mt][j] = 0.f;

  #pragma unroll
  for (int mt = 0; mt < 2; ++mt) {
    #pragma unroll
    for (int nt = 0; nt < 4; ++nt) {
      #pragma unroll
      for (int j = 0; j < 8; ++j) {
        const int m = wave * 32 + mt * 16 + half8 + j;
        float v = acc[mt][nt][j];
        if (bias) v += bias[m];
        const size_t idx = ioBase + (size_t)m * HWSZ + n0 + nt * 16 + ncol;
        if (outF) outF[idx] = v + (residual ? residual[idx] : 0.f);
        else      outB[idx] = (__bf16)v;
        ssl[mt][j] += v * v;
      }
    }
  }

  if (sspart) {
    #pragma unroll
    for (int mt = 0; mt < 2; ++mt)
      #pragma unroll
      for (int j = 0; j < 8; ++j) {
        float s = ssl[mt][j];
        s += __shfl_xor(s, 1);
        s += __shfl_xor(s, 2);
        s += __shfl_xor(s, 4);
        s += __shfl_xor(s, 8);
        if ((lane & 15) == 0) {
          const int m = wave * 32 + mt * 16 + half8 + j;
          sspart[(size_t)(b * CDIM + m) * NTILES_PER_IMG + tile] = s;
        }
      }
  }
}

// ---------------------------------------------------------------------------
// Deterministic sumsq reduction: 144 tile partials -> per (b,channel) scalar
// ---------------------------------------------------------------------------
__global__ __launch_bounds__(256)
void ssreduce_kernel(const float* __restrict__ qpart, const float* __restrict__ kpart,
                     float* __restrict__ qss, float* __restrict__ kss)
{
  const int i = blockIdx.x * 256 + threadIdx.x;       // 0..4095
  const int r = i & (BATCH * CDIM - 1);
  const float* src = (i < BATCH * CDIM) ? qpart : kpart;
  float*       dst = (i < BATCH * CDIM) ? qss : kss;
  float s = 0.f;
  for (int t = 0; t < NTILES_PER_IMG; ++t)
    s += src[(size_t)r * NTILES_PER_IMG + t];
  dst[r] = s;
}

// ---------------------------------------------------------------------------
// Attention: logits = (q . k^T) / (||q|| ||k||) * temp, softmax over d.
// One block per (b,h); 4 waves each own a 16x16 tile of the 32x32 logits,
// accumulating a K=9216 bf16 dot product through WMMA.
// ---------------------------------------------------------------------------
__global__ __launch_bounds__(128)
void attn_kernel(const __bf16* __restrict__ qbf, const __bf16* __restrict__ kbf,
                 const float* __restrict__ qss, const float* __restrict__ kss,
                 const float* __restrict__ temp, float* __restrict__ attn)
{
  __shared__ float lg[32][33];
  const int blk = blockIdx.x;
  const int b = blk >> 3, h = blk & 7;
  const int tid = threadIdx.x, wv = tid >> 5, lane = tid & 31;
  const int mt = wv >> 1, nt = wv & 1;
  const int r = lane & 15;
  const int koff  = (lane < 16) ? 0 : 8;
  const int khalf = (lane >> 4) << 4;
  const __bf16* qrow = qbf + (size_t)(b * CDIM + h * 32 + mt * 16 + r) * HWSZ;
  const __bf16* krow = kbf + (size_t)(b * CDIM + h * 32 + nt * 16 + r) * HWSZ;

  v8f acc = zero8();
  #pragma unroll 4
  for (int nb = 0; nb < HWSZ; nb += 32) {
    FragAB a, bb;
    a.v    = *(const v16bf*)(qrow + nb + khalf);
    bb.v   = *(const v16bf*)(krow + nb + khalf);
    acc = wmma_bf16(a.v, bb.v, acc);
  }

  const float tpr = temp[h];
  const int d = nt * 16 + (lane & 15);
  const int half8 = (lane >> 4) << 3;
  const float sk = 1.f / fmaxf(sqrtf(kss[b * CDIM + h * 32 + d]), 1e-12f);
  #pragma unroll
  for (int j = 0; j < 8; ++j) {
    const int c = mt * 16 + half8 + j;
    const float sq = 1.f / fmaxf(sqrtf(qss[b * CDIM + h * 32 + c]), 1e-12f);
    lg[c][d] = acc[j] * sq * sk * tpr;
  }
  __syncthreads();

  if (tid < 32) {
    const int c = tid;
    float mx = -3.4e38f;
    #pragma unroll
    for (int dd = 0; dd < 32; ++dd) mx = fmaxf(mx, lg[c][dd]);
    float e[32];
    float sum = 0.f;
    #pragma unroll
    for (int dd = 0; dd < 32; ++dd) { e[dd] = __expf(lg[c][dd] - mx); sum += e[dd]; }
    const float inv = 1.f / sum;
    float* arow = attn + ((size_t)(b * HEADS + h) * 32 + c) * 32;
    #pragma unroll
    for (int dd = 0; dd < 32; ++dd) arow[dd] = e[dd] * inv;
  }
}

// ---------------------------------------------------------------------------
// M_b = W_proj @ blockdiag(attn_b)  -> bf16 [B,256,256]
// Mb[c][d] = sum_e Wproj[c, (d>>5)*32+e] * attn[b, d>>5, e, d&31]
// ---------------------------------------------------------------------------
__global__ __launch_bounds__(256)
void mbuild_kernel(const float* __restrict__ Wproj, const float* __restrict__ attn,
                   __bf16* __restrict__ Mb)
{
  const int blk = blockIdx.x;           // B*256
  const int b = blk >> 8, c = blk & 255;
  const int d = threadIdx.x;
  const int h = d >> 5, dd = d & 31;
  const float* wr = Wproj + (size_t)c * CDIM + h * 32;
  const float* at = attn + ((size_t)(b * HEADS + h) * 32) * 32 + dd;
  float s = 0.f;
  #pragma unroll
  for (int e = 0; e < 32; ++e) s += wr[e] * at[e * 32];
  Mb[((size_t)b * CDIM + c) * CDIM + d] = (__bf16)s;
}

// ---------------------------------------------------------------------------
extern "C" void kernel_launch(void* const* d_in, const int* in_sizes, int n_in,
                              void* d_out, int out_size, void* d_ws, size_t ws_size,
                              hipStream_t stream)
{
  (void)in_sizes; (void)n_in; (void)out_size; (void)ws_size;
  const float* x     = (const float*)d_in[0];
  const float* y     = (const float*)d_in[1];
  const float* z     = (const float*)d_in[2];
  const float* Wqr   = (const float*)d_in[3];
  const float* bqr   = (const float*)d_in[4];
  const float* Wqd   = (const float*)d_in[5];
  const float* bqd   = (const float*)d_in[6];
  const float* Wkf   = (const float*)d_in[7];
  const float* bkf   = (const float*)d_in[8];
  const float* Wvf   = (const float*)d_in[9];
  const float* bvf   = (const float*)d_in[10];
  const float* Wcat  = (const float*)d_in[11];
  const float* bcat  = (const float*)d_in[12];
  const float* Wproj = (const float*)d_in[13];
  const float* bproj = (const float*)d_in[14];
  const float* temp  = (const float*)d_in[15];

  char* ws = (char*)d_ws;
  size_t off = 0;
  auto take = [&](size_t bytes) -> void* {
    size_t o = (off + 255) & ~(size_t)255;
    off = o + bytes;
    return (void*)(ws + o);
  };

  const size_t tenB = (size_t)BATCH * CDIM * HWSZ;
  __bf16* qbf  = (__bf16*)take(tenB * 2);
  __bf16* kbf  = (__bf16*)take(tenB * 2);
  __bf16* vbf  = (__bf16*)take(tenB * 2);
  __bf16* Wqx  = (__bf16*)take((size_t)CDIM * CDIM * 2);
  __bf16* Wqy  = (__bf16*)take((size_t)CDIM * CDIM * 2);
  __bf16* Wkb  = (__bf16*)take((size_t)CDIM * CDIM * 2);
  __bf16* Wvb  = (__bf16*)take((size_t)CDIM * CDIM * 2);
  float*  bq   = (float*) take((size_t)CDIM * 4);
  float* qpart = (float*) take((size_t)BATCH * CDIM * NTILES_PER_IMG * 4);
  float* kpart = (float*) take((size_t)BATCH * CDIM * NTILES_PER_IMG * 4);
  float*  qss  = (float*) take((size_t)BATCH * CDIM * 4);
  float*  kss  = (float*) take((size_t)BATCH * CDIM * 4);
  float* attnw = (float*) take((size_t)BATCH * HEADS * 32 * 32 * 4);
  __bf16* Mb   = (__bf16*)take((size_t)BATCH * CDIM * CDIM * 2);

  wprep_kernel<<<dim3(CDIM), dim3(CDIM), 0, stream>>>(
      Wqr, bqr, Wqd, bqd, Wkf, Wvf, Wcat, bcat, Wqx, Wqy, Wkb, Wvb, bq);

  const dim3 ggrid(BATCH * NTILES_PER_IMG);
  // q = Wqx@x + Wqy@y + bq        (bf16 out, sumsq partials)
  gemm256_kernel<<<ggrid, 256, 0, stream>>>(
      Wqx, Wqy, x, nullptr, y, bq, qbf, nullptr, qpart, nullptr, 0);
  // k = Wkf@z + bkf               (bf16 out, sumsq partials)
  gemm256_kernel<<<ggrid, 256, 0, stream>>>(
      Wkb, nullptr, z, nullptr, nullptr, bkf, kbf, nullptr, kpart, nullptr, 0);
  // v = Wvf@z + bvf               (bf16 out)
  gemm256_kernel<<<ggrid, 256, 0, stream>>>(
      Wvb, nullptr, z, nullptr, nullptr, bvf, vbf, nullptr, nullptr, nullptr, 0);

  ssreduce_kernel<<<dim3(16), dim3(256), 0, stream>>>(qpart, kpart, qss, kss);

  attn_kernel<<<dim3(BATCH * HEADS), dim3(128), 0, stream>>>(
      qbf, kbf, qss, kss, temp, attnw);

  mbuild_kernel<<<dim3(BATCH * CDIM), dim3(256), 0, stream>>>(Wproj, attnw, Mb);

  // out = Mb[b]@v + bproj + z     (fp32 out with residual)
  gemm256_kernel<<<ggrid, 256, 0, stream>>>(
      Mb, nullptr, nullptr, vbf, nullptr, bproj, nullptr, (float*)d_out, nullptr, z, 1);
}
